// ConditionByGraphEmbedding_12953621365081
// MI455X (gfx1250) — matrix operand order repaired and checked by reading, not compiled, not run on hardware
//
#include <hip/hip_runtime.h>
#include <hip/hip_bf16.h>
#include <stdint.h>

typedef __attribute__((ext_vector_type(16))) _Float16 v16h;
typedef __attribute__((ext_vector_type(8)))  _Float16 v8h;
typedef __attribute__((ext_vector_type(8)))  float    v8f;

#define B_SCENES 2048
#define NPER 16
#define EPER 240
#define NNODES (B_SCENES * NPER)
#define NEDGES (B_SCENES * EPER)
#define WSLOT 24576   // halves per transposed-weight slot (max 64*384)

__device__ __forceinline__ int lane_() { return threadIdx.x & 31; }
__device__ __forceinline__ int wave_() { return threadIdx.x >> 5; }

__device__ __forceinline__ v8f wmma16(v16h a, v16h b, v8f c) {
  return __builtin_amdgcn_wmma_f32_16x16x32_f16(false, a, false, b, (short)0, c, false, false);
}

// B fragment: WT is f16 [Npad][K] row-major (K contiguous). Lane l holds column
// n = l%16; halves 0..7 = K kf*32 + (l>=16 ? 8 : 0) + 0..7, halves 8..15 = that + 16.
__device__ __forceinline__ v16h loadB(const _Float16* WT, int K, int kf, int nt) {
  int l = lane_();
  return fragA_h(WT + (size_t)(nt * 16 + (l & 15)) * K + kf * 32 + ((l & 16) >> 1));
}

// A fragment builders. Caller passes p = &row[kbase + ((l&16)>>1)]; layout:
// halves 0..7 = K+0..7, halves 8..15 = K+16..23.
__device__ __forceinline__ v16h fragA_h(const _Float16* p) {
  v8h lo = *(const v8h*)p;
  v8h hi = *(const v8h*)(p + 16);
  v16h r;
#pragma unroll
  for (int i = 0; i < 8; ++i) { r[i] = lo[i]; r[i + 8] = hi[i]; }
  return r;
}
__device__ __forceinline__ v16h fragA_f(const float* p) {
  v16h r;
#pragma unroll
  for (int i = 0; i < 8; ++i) { r[i] = (_Float16)p[i]; r[i + 8] = (_Float16)p[i + 16]; }
  return r;
}
__device__ __forceinline__ v16h fragA_f_relu(const float* p) {
  v16h r;
#pragma unroll
  for (int i = 0; i < 8; ++i) {
    r[i]     = (_Float16)fmaxf(p[i], 0.f);
    r[i + 8] = (_Float16)fmaxf(p[i + 16], 0.f);
  }
  return r;
}
__device__ __forceinline__ v16h fragA_diff(const float* pd, const float* ps) {
  v16h r;
#pragma unroll
  for (int i = 0; i < 8; ++i) {
    r[i]     = (_Float16)(pd[i]      - ps[i]);
    r[i + 8] = (_Float16)(pd[i + 16] - ps[i + 16]);
  }
  return r;
}

__device__ __forceinline__ v8f accBias(const float* b, int nt, int nvalid) {
  int n = nt * 16 + (lane_() & 15);
  float bv = (n < nvalid) ? b[n] : 0.f;
  v8f a;
#pragma unroll
  for (int i = 0; i < 8; ++i) a[i] = bv;
  return a;
}

// C/D layout: VGPR r holds (M = r + 8*(l>=16), N = l%16). Store to per-wave
// 16x64 f16 LDS tile (A-layout source for the next layer).
__device__ __forceinline__ void storeAct(_Float16* act, int nt, v8f d, bool relu) {
  int l  = lane_();
  int m0 = (l & 16) >> 1;
  int n  = nt * 16 + (l & 15);
#pragma unroll
  for (int r = 0; r < 8; ++r) {
    float v = d[r];
    if (relu) v = fmaxf(v, 0.f);
    act[(m0 + r) * 64 + n] = (_Float16)v;
  }
}

template <int NKF, typename FA>
__device__ __forceinline__ void mlpLayer1(FA buildA, const int* kfs,
                                          const _Float16* WT1, int K1,
                                          const float* b1, _Float16* act) {
  v16h a[NKF];
#pragma unroll
  for (int i = 0; i < NKF; ++i) a[i] = buildA(i);
#pragma unroll
  for (int nt = 0; nt < 4; ++nt) {
    v8f acc = accBias(b1, nt, 64);
#pragma unroll
    for (int i = 0; i < NKF; ++i)
      acc = wmma16(a[i], loadB(WT1, K1, kfs[i], nt), acc);
    storeAct(act, nt, acc, true);
  }
}

__device__ __forceinline__ void mlpTail(_Float16* act,
                                        const _Float16* WT2, const float* b2,
                                        const _Float16* WT3, const float* b3,
                                        int n3tiles, int n3valid, v8f* dout) {
  int l = lane_();
  const _Float16* ar = act + (l & 15) * 64 + ((l & 16) >> 1);
  v16h a0 = fragA_h(ar);
  v16h a1 = fragA_h(ar + 32);
#pragma unroll
  for (int nt = 0; nt < 4; ++nt) {
    v8f acc = accBias(b2, nt, 64);
    acc = wmma16(a0, loadB(WT2, 64, 0, nt), acc);
    acc = wmma16(a1, loadB(WT2, 64, 1, nt), acc);
    storeAct(act, nt, acc, true);
  }
  a0 = fragA_h(ar);
  a1 = fragA_h(ar + 32);
  for (int nt = 0; nt < n3tiles; ++nt) {
    v8f acc = accBias(b3, nt, n3valid);
    acc = wmma16(a0, loadB(WT3, 64, 0, nt), acc);
    acc = wmma16(a1, loadB(WT3, 64, 1, nt), acc);
    dout[nt] = acc;
  }
}

// ---------------- Edge model (fused edge MLP + dst-mean + scene edge-mean) ---
template <bool G2>
__global__ void __launch_bounds__(128) meta_edge_kernel(
    const float* xbuf, _Float16* ebuf, const float* ubuf, const float* u1pre,
    const _Float16* WT1, const float* b1, const _Float16* WT2, const float* b2,
    const _Float16* WT3, const float* b3, float* aggOut, float* esumOut) {
  __shared__ float xs[NPER * 64];
  __shared__ float aggRaw[NPER * 64];
  __shared__ __align__(32) _Float16 act[4][16 * 64];
  int scene = blockIdx.x;
  for (int i = threadIdx.x; i < NPER * 64; i += blockDim.x) {
    xs[i] = xbuf[(size_t)scene * NPER * 64 + i];
    aggRaw[i] = 0.f;
  }
  __syncthreads();
  int w = wave_(), l = lane_();
  int sub = (l & 16) >> 1;
  const float* uS  = ubuf + (size_t)scene * 64;
  const float* upS = G2 ? (u1pre + (size_t)scene * 64) : nullptr;
  const int K1 = G2 ? 384 : 192;
  const int kfsG1[6]  = {0, 1, 2, 3, 4, 5};
  const int kfsG2[10] = {0, 1, 2, 3, 4, 5, 8, 9, 10, 11};  // skip zero K-block 192..255

  for (int mt = w; mt < 15; mt += 4) {
    int m  = mt * 16 + (l & 15);          // A-layout row for this lane
    int si = m / 15, jj = m % 15;
    int di = jj + (jj >= si ? 1 : 0);
    const _Float16* er = ebuf + ((size_t)scene * EPER + m) * 64;
    v8f dout[4];
    if (!G2) {
      auto bA = [&](int i) -> v16h {
        switch (i) {
          case 0: case 1: return fragA_h(er + i * 32 + sub);
          case 2: case 3: return fragA_diff(&xs[di * 64 + (i - 2) * 32 + sub],
                                            &xs[si * 64 + (i - 2) * 32 + sub]);
          default:        return fragA_f(uS + (i - 4) * 32 + sub);
        }
      };
      mlpLayer1<6>(bA, kfsG1, WT1, K1, b1, act[w]);
    } else {
      auto bA = [&](int i) -> v16h {
        switch (i) {
          case 0: case 1: return fragA_h(er + i * 32 + sub);
          case 2: case 3: return fragA_f(upS + (i - 2) * 32 + sub);
          case 4: case 5: return fragA_diff(&xs[di * 64 + (i - 4) * 32 + sub],
                                            &xs[si * 64 + (i - 4) * 32 + sub]);
          case 6: case 7: return fragA_f(uS + (i - 6) * 32 + sub);
          default:        return fragA_f(upS + (i - 8) * 32 + sub);
        }
      };
      mlpLayer1<10>(bA, kfsG2, WT1, K1, b1, act[w]);
    }
    mlpTail(act[w], WT2, b2, WT3, b3, 4, 64, dout);
    int m0 = mt * 16 + sub;                // D-layout row base for this lane
#pragma unroll
    for (int nt = 0; nt < 4; ++nt) {
      int n = nt * 16 + (l & 15);
#pragma unroll
      for (int r = 0; r < 8; ++r) {
        float v = dout[nt][r];
        int me = m0 + r;
        ebuf[((size_t)scene * EPER + me) * 64 + n] = (_Float16)v;
        int ei = me / 15, ej = me % 15;
        int dn = ej + (ej >= ei ? 1 : 0);
        atomicAdd(&aggRaw[dn * 64 + n], v);
      }
    }
  }
  __syncthreads();
  for (int i = threadIdx.x; i < NPER * 64; i += blockDim.x)
    aggOut[(size_t)scene * NPER * 64 + i] = aggRaw[i] * (1.f / 15.f);
  if (threadIdx.x < 64) {
    float s = 0.f;
#pragma unroll
    for (int j = 0; j < NPER; ++j) s += aggRaw[j * 64 + threadIdx.x];
    esumOut[(size_t)scene * 64 + threadIdx.x] = s * (1.f / 240.f);  // all 240 edges
  }
}

// ---------------- Node model (wave per scene: 16 nodes = one M-tile) --------
template <bool G2>
__global__ void __launch_bounds__(256) meta_node_kernel(
    float* xbuf, const float* agg, const float* ubuf, const float* u1pre,
    const _Float16* WT1, const float* b1, const _Float16* WT2, const float* b2,
    const _Float16* WT3, const float* b3, float* xsumOut) {
  __shared__ __align__(32) _Float16 act[8][16 * 64];
  int w = wave_(), l = lane_();
  int scene = blockIdx.x * 8 + w;
  int sub = (l & 16) >> 1;
  const float* xr  = xbuf + ((size_t)scene * 16 + (l & 15)) * 64;
  const float* ar  = agg  + ((size_t)scene * 16 + (l & 15)) * 64;
  const float* uS  = ubuf + (size_t)scene * 64;
  const float* upS = G2 ? (u1pre + (size_t)scene * 64) : nullptr;
  const int kfs[10] = {0, 1, 2, 3, 4, 5, 6, 7, 8, 9};
  v8f dout[4];
  if (!G2) {
    auto bA = [&](int i) -> v16h {
      switch (i) {
        case 0: case 1: return fragA_f(xr + i * 32 + sub);
        case 2: case 3: return fragA_f(ar + (i - 2) * 32 + sub);
        default:        return fragA_f(uS + (i - 4) * 32 + sub);
      }
    };
    mlpLayer1<6>(bA, kfs, WT1, 192, b1, act[w]);
  } else {
    auto bA = [&](int i) -> v16h {
      switch (i) {
        case 0: case 1: return fragA_f(xr + i * 32 + sub);
        case 2: case 3: return fragA_f(upS + (i - 2) * 32 + sub);
        case 4: case 5: return fragA_f(ar + (i - 4) * 32 + sub);
        case 6: case 7: return fragA_f(uS + (i - 6) * 32 + sub);
        default:        return fragA_f(upS + (i - 8) * 32 + sub);
      }
    };
    mlpLayer1<10>(bA, kfs, WT1, 320, b1, act[w]);
  }
  mlpTail(act[w], WT2, b2, WT3, b3, 4, 64, dout);
#pragma unroll
  for (int nt = 0; nt < 4; ++nt) {
    int n = nt * 16 + (l & 15);
    float cs = 0.f;
#pragma unroll
    for (int r = 0; r < 8; ++r) {
      float v = dout[nt][r];
      xbuf[((size_t)scene * 16 + sub + r) * 64 + n] = v;
      cs += v;
    }
    cs += __shfl_xor(cs, 16);
    if (l < 16) xsumOut[(size_t)scene * 64 + n] = cs * (1.f / 16.f);
  }
}

// ---------------- Global model (wave per 16-scene row tile) -----------------
template <bool G2>
__global__ void __launch_bounds__(256) meta_global_kernel(
    const float* s0, const float* s1, const float* ubuf, const float* u1pre,
    const _Float16* WT1, const float* b1, const _Float16* WT2, const float* b2,
    const _Float16* WT3, const float* b3, float* out) {
  __shared__ __align__(32) _Float16 act[8][16 * 64];
  int w = wave_(), l = lane_();
  int tile = blockIdx.x * 8 + w;
  int sub = (l & 16) >> 1;
  int row = tile * 16 + (l & 15);
  const float* p0 = s0 + (size_t)row * 64;
  const float* p1 = s1 + (size_t)row * 64;
  const float* pu = ubuf + (size_t)row * 64;
  const float* pp = G2 ? (u1pre + (size_t)row * 64) : nullptr;
  const int kfs[8] = {0, 1, 2, 3, 4, 5, 6, 7};
  v8f dout[4];
  if (!G2) {
    auto bA = [&](int i) -> v16h {
      switch (i) {
        case 0: case 1: return fragA_f(p0 + i * 32 + sub);
        case 2: case 3: return fragA_f(p1 + (i - 2) * 32 + sub);
        default:        return fragA_f(pu + (i - 4) * 32 + sub);
      }
    };
    mlpLayer1<6>(bA, kfs, WT1, 192, b1, act[w]);
  } else {
    auto bA = [&](int i) -> v16h {
      switch (i) {
        case 0: case 1: return fragA_f(p0 + i * 32 + sub);
        case 2: case 3: return fragA_f(p1 + (i - 2) * 32 + sub);
        case 4: case 5: return fragA_f(pu + (i - 4) * 32 + sub);
        default:        return fragA_f(pp + (i - 6) * 32 + sub);
      }
    };
    mlpLayer1<8>(bA, kfs, WT1, 256, b1, act[w]);
  }
  mlpTail(act[w], WT2, b2, WT3, b3, 4, 64, dout);
#pragma unroll
  for (int nt = 0; nt < 4; ++nt) {
    int n = nt * 16 + (l & 15);
#pragma unroll
    for (int r = 0; r < 8; ++r)
      out[((size_t)tile * 16 + sub + r) * 64 + n] = dout[nt][r];
  }
}

// ---------------- resize_u (relu+linear) + readout MLP ----------------------
__global__ void __launch_bounds__(256) finish_kernel(
    const float* u1pre, float* u1out, const float* u2,
    const _Float16* WTr, const float* br,
    const _Float16* WTo1, const float* bo1,
    const _Float16* WTo2, const float* bo2,
    const _Float16* WTo3, const float* bo3,
    float* outp, int step) {
  __shared__ __align__(32) _Float16 act[8][16 * 64];
  int w = wave_(), l = lane_();
  int tile = blockIdx.x * 8 + w;
  int sub = (l & 16) >> 1;
  int row = tile * 16 + (l & 15);
  {
    const float* pr = u1pre + (size_t)row * 64;
    v16h a0 = fragA_f_relu(pr + sub);
    v16h a1 = fragA_f_relu(pr + 32 + sub);
#pragma unroll
    for (int nt = 0; nt < 4; ++nt) {
      v8f acc = accBias(br, nt, 64);
      acc = wmma16(a0, loadB(WTr, 64, 0, nt), acc);
      acc = wmma16(a1, loadB(WTr, 64, 1, nt), acc);
      int n = nt * 16 + (l & 15);
#pragma unroll
      for (int r = 0; r < 8; ++r)
        u1out[((size_t)tile * 16 + sub + r) * 64 + n] = acc[r];
    }
  }
  const float* pu = u2 + (size_t)row * 64;
  const int kfs[2] = {0, 1};
  auto bA = [&](int i) -> v16h { return fragA_f(pu + i * 32 + sub); };
  mlpLayer1<2>(bA, kfs, WTo1, 64, bo1, act[w]);
  v8f dout[4];
  mlpTail(act[w], WTo2, bo2, WTo3, bo3, 1, 2, dout);
  int n = l & 15;
  if (n < 2) {
#pragma unroll
    for (int r = 0; r < 8; ++r)
      outp[(size_t)step * (B_SCENES * 2) + ((size_t)tile * 16 + sub + r) * 2 + n] =
          dout[0][r];
  }
}

// ---------------- prep: fp32 [K][N] weights -> f16 [Npad][K] transposed -----
struct WDesc { const float* src; int K; int Npad; int Nvalid; };
struct WPack { WDesc d[22]; };

__global__ void prep_weights_kernel(WPack p, _Float16* wt) {
  int mi = blockIdx.y;
  WDesc d = p.d[mi];
  _Float16* dst = wt + (size_t)mi * WSLOT;
  int total = d.Npad * d.K;
  for (int i = blockIdx.x * blockDim.x + threadIdx.x; i < total;
       i += gridDim.x * blockDim.x) {
    int n = i / d.K, k = i - n * d.K;
    float v = (n < d.Nvalid) ? d.src[(size_t)k * d.Nvalid + n] : 0.f;
    dst[i] = (_Float16)v;
  }
}

__global__ void cvt_kernel(const float* s, _Float16* d, size_t n) {
  for (size_t i = blockIdx.x * (size_t)blockDim.x + threadIdx.x; i < n;
       i += (size_t)gridDim.x * blockDim.x)
    d[i] = (_Float16)s[i];
}

// ---------------------------------------------------------------------------
extern "C" void kernel_launch(void* const* d_in, const int* in_sizes, int n_in,
                              void* d_out, int out_size, void* d_ws, size_t ws_size,
                              hipStream_t stream) {
  (void)in_sizes; (void)n_in; (void)out_size;
  char* ws = (char*)d_ws;
  size_t off = 0;
  auto alloc = [&](size_t bytes) -> char* {
    char* p = ws + off;
    off = (off + bytes + 255) & ~(size_t)255;
    return p;
  };
  _Float16* e1h = (_Float16*)alloc((size_t)NEDGES * 64 * 2);
  _Float16* e2h = (_Float16*)alloc((size_t)NEDGES * 64 * 2);
  float* x1   = (float*)alloc((size_t)NNODES * 64 * 4);
  float* x2   = (float*)alloc((size_t)NNODES * 64 * 4);
  float* u1   = (float*)alloc((size_t)B_SCENES * 64 * 4);
  float* u2   = (float*)alloc((size_t)B_SCENES * 64 * 4);
  float* u1p  = (float*)alloc((size_t)B_SCENES * 64 * 4);
  float* agg1 = (float*)alloc((size_t)NNODES * 64 * 4);
  float* agg2 = (float*)alloc((size_t)NNODES * 64 * 4);
  float* es1  = (float*)alloc((size_t)B_SCENES * 64 * 4);
  float* xs1  = (float*)alloc((size_t)B_SCENES * 64 * 4);
  float* es2  = (float*)alloc((size_t)B_SCENES * 64 * 4);
  float* xs2  = (float*)alloc((size_t)B_SCENES * 64 * 4);
  _Float16* wt = (_Float16*)alloc((size_t)22 * WSLOT * 2);
  if (ws_size < off) return;

  auto F = [&](int i) { return (const float*)d_in[i]; };
  auto WTs = [&](int s) { return (const _Float16*)(wt + (size_t)s * WSLOT); };

  WPack p;
  auto setW = [&](int slot, int idx, int K, int Npad, int Nv) {
    p.d[slot].src = F(idx); p.d[slot].K = K; p.d[slot].Npad = Npad; p.d[slot].Nvalid = Nv;
  };
  setW(0, 10, 192, 64, 64);  setW(1, 12, 64, 64, 64);  setW(2, 14, 64, 64, 64);   // edge_q
  setW(3, 16, 192, 64, 64);  setW(4, 18, 64, 64, 64);  setW(5, 20, 64, 64, 64);   // node_q
  setW(6, 22, 192, 64, 64);  setW(7, 24, 64, 64, 64);  setW(8, 26, 64, 64, 64);   // global_q
  setW(9, 28, 384, 64, 64);  setW(10, 30, 64, 64, 64); setW(11, 32, 64, 64, 64);  // edge_w
  setW(12, 34, 320, 64, 64); setW(13, 36, 64, 64, 64); setW(14, 38, 64, 64, 64);  // node_w
  setW(15, 40, 256, 64, 64); setW(16, 42, 64, 64, 64); setW(17, 44, 64, 64, 64);  // global_w
  setW(18, 46, 64, 64, 64);                                                       // resize
  setW(19, 48, 64, 64, 64);  setW(20, 50, 64, 64, 64); setW(21, 52, 64, 16, 2);   // readout

  hipMemcpyAsync(x1, F(0), (size_t)NNODES * 64 * 4, hipMemcpyDeviceToDevice, stream);
  hipMemcpyAsync(x2, F(3), (size_t)NNODES * 64 * 4, hipMemcpyDeviceToDevice, stream);
  hipMemcpyAsync(u1, F(2), (size_t)B_SCENES * 64 * 4, hipMemcpyDeviceToDevice, stream);
  hipMemcpyAsync(u2, F(5), (size_t)B_SCENES * 64 * 4, hipMemcpyDeviceToDevice, stream);
  prep_weights_kernel<<<dim3(8, 22), 256, 0, stream>>>(p, wt);
  cvt_kernel<<<2048, 256, 0, stream>>>(F(1), e1h, (size_t)NEDGES * 64);
  cvt_kernel<<<2048, 256, 0, stream>>>(F(4), e2h, (size_t)NEDGES * 64);

  for (int step = 0; step < 3; ++step) {
    meta_edge_kernel<false><<<B_SCENES, 128, 0, stream>>>(
        x1, e1h, u1, nullptr, WTs(0), F(11), WTs(1), F(13), WTs(2), F(15), agg1, es1);
    meta_node_kernel<false><<<B_SCENES / 8, 256, 0, stream>>>(
        x1, agg1, u1, nullptr, WTs(3), F(17), WTs(4), F(19), WTs(5), F(21), xs1);
    meta_global_kernel<false><<<B_SCENES / 128, 256, 0, stream>>>(
        es1, xs1, u1, nullptr, WTs(6), F(23), WTs(7), F(25), WTs(8), F(27), u1p);
    meta_edge_kernel<true><<<B_SCENES, 128, 0, stream>>>(
        x2, e2h, u2, u1p, WTs(9), F(29), WTs(10), F(31), WTs(11), F(33), agg2, es2);
    meta_node_kernel<true><<<B_SCENES / 8, 256, 0, stream>>>(
        x2, agg2, u2, u1p, WTs(12), F(35), WTs(13), F(37), WTs(14), F(39), xs2);
    meta_global_kernel<true><<<B_SCENES / 128, 256, 0, stream>>>(
        es2, xs2, u2, u1p, WTs(15), F(41), WTs(16), F(43), WTs(17), F(45), u2);
    finish_kernel<<<B_SCENES / 128, 256, 0, stream>>>(
        u1p, u1, u2, WTs(18), F(47), WTs(19), F(49), WTs(20), F(51), WTs(21), F(53),
        (float*)d_out, step);
  }
}
